// PolicyNetwork_2542620639964
// MI455X (gfx1250) — matrix operand order repaired and checked by reading, not compiled, not run on hardware
//
#include <hip/hip_runtime.h>
#include <hip/hip_bf16.h>

// ---------------------------------------------------------------------------
// PolicyNetwork fused pipeline for MI455X (gfx1250, wave32, WMMA)
//
// H=256, HEADS=8 (d=32), N = 131072 rows, ~86 GFLOP total. All GEMM work runs
// on v_wmma_f32_16x16x32_f16. Weights are converted to f16 ONCE (L2-resident,
// ~0.9 MB) so the hot loop has zero cvt instructions; keys are materialized
// once in f16 (64 MB) and re-read twice; k/v are never materialized
// (v is folded: attn = wv @ (softmax-weighted keys sum) + bv).
// ---------------------------------------------------------------------------

#define HDIM 256
#define NMAP 65536
#define NTOT 131072

// f16 weight cache element offsets inside ws
#define W1M_OFF 0         // map_w1  256x512
#define W1S_OFF 131072    // sch_w1  256x512
#define W2M_OFF 262144    // map_w2  256x256
#define W2S_OFF 327680    // sch_w2  256x256
#define WK_OFF  393216    // wk = attn_in_w[256:512]  256x256
#define WH_TOTAL 458752

typedef __attribute__((ext_vector_type(16))) _Float16 v16h;
typedef __attribute__((ext_vector_type(8)))  _Float16 v8h;
typedef __attribute__((ext_vector_type(8)))  float    v8f;

// A-fragment (16x32 f16, MxK) from row-major f16 LDS tile.
// Lane L holds row M=L&15; slot s -> K = 16*(L>>4)+s (matches load_b_h).
// Two ds_load_b128.
__device__ __forceinline__ v16h load_a_lds(const _Float16* base, int stride,
                                           int kbase, int lane) {
  const _Float16* p = base + (lane & 15) * stride + kbase + ((lane >> 4) << 4);
  v8h lo = *(const v8h*)p;
  v8h hi = *(const v8h*)(p + 8);
  return __builtin_shufflevector(lo, hi, 0, 1, 2, 3, 4, 5, 6, 7,
                                 8, 9, 10, 11, 12, 13, 14, 15);
}

// B-fragment (32x16 f16, KxN) where B[k][n] = Wh[nbase+n][kbase+k], Wh f16
// row-major. Lane L holds column N=L&15; slot s -> K = 16*(L>>4)+s.
// One 32-byte contiguous vector load per lane, no conversion.
__device__ __forceinline__ v16h load_b_h(const _Float16* Wh, int Kdim,
                                         int nbase, int kbase, int lane) {
  const _Float16* p = Wh + (size_t)(nbase + (lane & 15)) * Kdim + kbase +
                      ((lane >> 4) << 4);
  return *(const v16h*)p;
}

// ---------------------------------------------------------------------------
// P0: one-time fp32 -> f16 weight conversion into ws (L2-resident cache).
// ---------------------------------------------------------------------------
__global__ __launch_bounds__(256) void convert_weights_kernel(
    const float* __restrict__ map_w1, const float* __restrict__ sch_w1,
    const float* __restrict__ map_w2, const float* __restrict__ sch_w2,
    const float* __restrict__ attn_in_w, _Float16* __restrict__ wh) {
  const int i = blockIdx.x * 256 + threadIdx.x;
  if (i >= WH_TOTAL) return;
  float v;
  if (i < W1S_OFF)      v = map_w1[i];
  else if (i < W2M_OFF) v = sch_w1[i - W1S_OFF];
  else if (i < W2S_OFF) v = map_w2[i - W2M_OFF];
  else if (i < WK_OFF)  v = sch_w2[i - W2S_OFF];
  else                  v = attn_in_w[(size_t)HDIM * HDIM + (i - WK_OFF)];
  wh[i] = (_Float16)v;
}

// ---------------------------------------------------------------------------
// K0: g = mean(all_emb); query = mlp(g); q = query @ wq^T + bq  (one block)
// ---------------------------------------------------------------------------
__global__ __launch_bounds__(256) void query_kernel(
    const float* __restrict__ all_emb,
    const float* __restrict__ qg_w1, const float* __restrict__ qg_b1,
    const float* __restrict__ qg_w2, const float* __restrict__ qg_b2,
    const float* __restrict__ attn_in_w, const float* __restrict__ attn_in_b,
    float* __restrict__ q_ws) {
  __shared__ float g[HDIM], hid[HDIM], query[HDIM];
  const int t = threadIdx.x;
  float acc = 0.f;
  for (int r = 0; r < 8192; ++r) acc += all_emb[(size_t)r * HDIM + t];
  g[t] = acc * (1.0f / 8192.0f);
  __syncthreads();
  float a1 = qg_b1[t];
  for (int j = 0; j < HDIM; ++j) a1 += g[j] * qg_w1[(size_t)t * HDIM + j];
  hid[t] = a1 > 0.f ? a1 : 0.f;
  __syncthreads();
  float a2 = qg_b2[t];
  for (int i = 0; i < HDIM; ++i) a2 += hid[i] * qg_w2[(size_t)t * HDIM + i];
  query[t] = a2;
  __syncthreads();
  float a3 = attn_in_b[t];  // wq = attn_in_w[0:256]
  for (int j = 0; j < HDIM; ++j) a3 += query[j] * attn_in_w[(size_t)t * HDIM + j];
  q_ws[t] = a3;
}

// ---------------------------------------------------------------------------
// K1: fused gather -> MLP(l1,l2) -> keys(f16) -> k-projection -> scores.
// One block = 16 rows x 256 cols, 8 waves. Each wave owns two 16-col tiles
// and keeps two accumulators so every A-fragment feeds two back-to-back WMMAs.
// ---------------------------------------------------------------------------
__global__ __launch_bounds__(256) void keys_scores_kernel(
    const float* __restrict__ qubit_emb, const float* __restrict__ qpu_emb,
    const float* __restrict__ gate_emb,  const float* __restrict__ time_table,
    const float* __restrict__ map_b1, const float* __restrict__ map_b2,
    const float* __restrict__ sch_b1, const float* __restrict__ sch_b2,
    const float* __restrict__ attn_in_b,
    const _Float16* __restrict__ wh,
    const int* __restrict__ map_qubit, const int* __restrict__ map_qpu,
    const int* __restrict__ sched_gate, const int* __restrict__ sched_time,
    const float* __restrict__ q_ws,
    _Float16* __restrict__ keys_ws, float* __restrict__ scores_ws) {
  __shared__ _Float16 x_lds[16 * 512];      // gathered input tile (f16)
  __shared__ _Float16 hid_lds[16 * HDIM];   // relu hidden (f16)
  __shared__ _Float16 keys_lds[16 * HDIM];  // keys tile (f16)
  __shared__ float    kproj_lds[16 * HDIM]; // k-projection (f32)

  const int tid = threadIdx.x;
  const int lane = tid & 31;
  const int wave = tid >> 5;
  const int half = lane >> 4;
  const int nlane = lane & 15;
  const int n0 = blockIdx.x * 16;
  const bool is_map = (n0 < NMAP);  // NM multiple of 16: tiles don't straddle

  // ---- gather 16x512 embedding rows, fp32 -> f16 ----
  for (int i = tid; i < 16 * 512; i += 256) {
    const int m = i >> 9;
    const int c = i & 511;
    const int n = n0 + m;
    float v;
    if (is_map) {
      if (c < HDIM) v = qubit_emb[(size_t)map_qubit[n] * HDIM + c];
      else          v = qpu_emb[(size_t)map_qpu[n] * HDIM + (c - HDIM)];
    } else {
      const int ns = n - NMAP;
      if (c < HDIM) v = gate_emb[(size_t)sched_gate[ns] * HDIM + c];
      else          v = time_table[(size_t)sched_time[ns] * HDIM + (c - HDIM)];
    }
    x_lds[i] = (_Float16)v;
  }
  __syncthreads();

  const _Float16* w1h = wh + (is_map ? W1M_OFF : W1S_OFF);
  const _Float16* w2h = wh + (is_map ? W2M_OFF : W2S_OFF);
  const _Float16* wkh = wh + WK_OFF;
  const float* b1 = is_map ? map_b1 : sch_b1;
  const float* b2 = is_map ? map_b2 : sch_b2;

  const int nb0 = wave * 32;       // wave's first 16-col tile
  const int nb1 = wave * 32 + 16;  // wave's second 16-col tile

  // ---- layer 1: hidden = relu(x @ w1^T + b1), K=512 (16 K-steps) ----
  {
    v8f acc0 = {}, acc1 = {};
    for (int kt = 0; kt < 16; ++kt) {
      v16h a  = load_a_lds(x_lds, 512, kt * 32, lane);
      v16h bb0 = load_b_h(w1h, 512, nb0, kt * 32, lane);
      v16h bb1 = load_b_h(w1h, 512, nb1, kt * 32, lane);
      if (kt + 1 < 16) {  // global_prefetch_b8 for next K-step's weights
        __builtin_prefetch((const void*)(w1h + (size_t)(nb0 + nlane) * 512 +
                                         (kt + 1) * 32 + (half << 4)), 0, 0);
      }
      acc0 = __builtin_amdgcn_wmma_f32_16x16x32_f16(false, a, false, bb0,
                                                    (short)0, acc0, false, false);
      acc1 = __builtin_amdgcn_wmma_f32_16x16x32_f16(false, a, false, bb1,
                                                    (short)0, acc1, false, false);
    }
#pragma unroll
    for (int r = 0; r < 8; ++r) {  // C/D layout: m = r + 8*half, n = lane&15
      const int m = half * 8 + r;
      float h0 = acc0[r] + b1[nb0 + nlane];
      float h1 = acc1[r] + b1[nb1 + nlane];
      hid_lds[m * HDIM + nb0 + nlane] = (_Float16)(h0 > 0.f ? h0 : 0.f);
      hid_lds[m * HDIM + nb1 + nlane] = (_Float16)(h1 > 0.f ? h1 : 0.f);
    }
  }
  __syncthreads();

  // ---- layer 2: keys = hidden @ w2^T + b2, K=256 (8 K-steps) ----
  {
    v8f acc0 = {}, acc1 = {};
    for (int kt = 0; kt < 8; ++kt) {
      v16h a  = load_a_lds(hid_lds, HDIM, kt * 32, lane);
      v16h bb0 = load_b_h(w2h, HDIM, nb0, kt * 32, lane);
      v16h bb1 = load_b_h(w2h, HDIM, nb1, kt * 32, lane);
      acc0 = __builtin_amdgcn_wmma_f32_16x16x32_f16(false, a, false, bb0,
                                                    (short)0, acc0, false, false);
      acc1 = __builtin_amdgcn_wmma_f32_16x16x32_f16(false, a, false, bb1,
                                                    (short)0, acc1, false, false);
    }
#pragma unroll
    for (int r = 0; r < 8; ++r) {
      const int m = half * 8 + r;
      keys_lds[m * HDIM + nb0 + nlane] = (_Float16)(acc0[r] + b2[nb0 + nlane]);
      keys_lds[m * HDIM + nb1 + nlane] = (_Float16)(acc1[r] + b2[nb1 + nlane]);
    }
  }
  __syncthreads();

  // ---- coalesced keys materialization: 32 contiguous bytes per thread ----
  {
    const int m = tid >> 4;            // 256 threads * 16 f16 = 16x256 tile
    const int col = (tid & 15) << 4;
    *(v16h*)(keys_ws + (size_t)(n0 + m) * HDIM + col) =
        *(const v16h*)(keys_lds + m * HDIM + col);
  }

  // ---- k-projection: k = keys @ wk^T + bk, wk = attn_in_w[256:512] ----
  {
    v8f acc0 = {}, acc1 = {};
    for (int kt = 0; kt < 8; ++kt) {
      v16h a  = load_a_lds(keys_lds, HDIM, kt * 32, lane);
      v16h bb0 = load_b_h(wkh, HDIM, nb0, kt * 32, lane);
      v16h bb1 = load_b_h(wkh, HDIM, nb1, kt * 32, lane);
      acc0 = __builtin_amdgcn_wmma_f32_16x16x32_f16(false, a, false, bb0,
                                                    (short)0, acc0, false, false);
      acc1 = __builtin_amdgcn_wmma_f32_16x16x32_f16(false, a, false, bb1,
                                                    (short)0, acc1, false, false);
    }
    const float* bk = attn_in_b + HDIM;
#pragma unroll
    for (int r = 0; r < 8; ++r) {
      const int m = half * 8 + r;
      kproj_lds[m * HDIM + nb0 + nlane] = acc0[r] + bk[nb0 + nlane];
      kproj_lds[m * HDIM + nb1 + nlane] = acc1[r] + bk[nb1 + nlane];
    }
  }
  __syncthreads();

  // ---- per-head scores: s[h][n] = (q_h . k_{n,h}) / sqrt(32) ----
  if (tid < 128) {
    const int m = tid >> 3, h = tid & 7;
    const float* qh = q_ws + h * 32;
    const float* kh = kproj_lds + m * HDIM + h * 32;
    float s = 0.f;
#pragma unroll
    for (int d = 0; d < 32; ++d) s += qh[d] * kh[d];
    scores_ws[(size_t)h * NTOT + (n0 + m)] = s * 0.17677669529663687f; // 1/sqrt(32)
  }
}

// ---------------------------------------------------------------------------
// K2: per-head softmax over n (8 blocks), emit normalized weights; zero u.
// ---------------------------------------------------------------------------
__global__ __launch_bounds__(256) void head_softmax_kernel(
    const float* __restrict__ scores_ws, float* __restrict__ stats_ws,
    float* __restrict__ wnorm_ws, float* __restrict__ u_ws) {
  __shared__ float red[256];
  const int h = blockIdx.x;
  const int t = threadIdx.x;
  const float* s = scores_ws + (size_t)h * NTOT;
  float mx = -3.0e38f;
  for (int n = t; n < NTOT; n += 256) mx = fmaxf(mx, s[n]);
  red[t] = mx; __syncthreads();
  for (int off = 128; off > 0; off >>= 1) {
    if (t < off) red[t] = fmaxf(red[t], red[t + off]);
    __syncthreads();
  }
  mx = red[0]; __syncthreads();
  float sum = 0.f;
  for (int n = t; n < NTOT; n += 256) sum += __expf(s[n] - mx);
  red[t] = sum; __syncthreads();
  for (int off = 128; off > 0; off >>= 1) {
    if (t < off) red[t] += red[t + off];
    __syncthreads();
  }
  sum = red[0];
  if (t == 0) { stats_ws[h] = mx; stats_ws[8 + h] = sum; }
  const float inv = 1.0f / sum;
  for (int n = t; n < NTOT; n += 256)
    wnorm_ws[(size_t)n * 8 + h] = __expf(s[n] - mx) * inv;
  u_ws[h * HDIM + t] = 0.f;  // init for K3's atomics
}

// ---------------------------------------------------------------------------
// K3: u[h][c] = sum_n w[h][n] * keys[n][c]   (v folded algebraically)
// ---------------------------------------------------------------------------
__global__ __launch_bounds__(256) void weighted_sum_kernel(
    const float* __restrict__ wnorm_ws, const _Float16* __restrict__ keys_ws,
    float* __restrict__ u_ws) {
  const int c = threadIdx.x;
  const int b = blockIdx.x;
  float acc[8];
#pragma unroll
  for (int h = 0; h < 8; ++h) acc[h] = 0.f;
  for (int r = 0; r < 256; ++r) {
    const int n = b * 256 + r;
    const float* wn = wnorm_ws + (size_t)n * 8;             // wave-uniform
    const float kf = (float)keys_ws[(size_t)n * HDIM + c];  // coalesced
#pragma unroll
    for (int h = 0; h < 8; ++h) acc[h] += wn[h] * kf;
  }
#pragma unroll
  for (int h = 0; h < 8; ++h) atomicAdd(&u_ws[h * HDIM + c], acc[h]);
}

// ---------------------------------------------------------------------------
// K4: attn = wv @ u + bv (softmax weights sum to 1), then attn_out matvec.
// ---------------------------------------------------------------------------
__global__ __launch_bounds__(256) void attn_out_kernel(
    const float* __restrict__ attn_in_w, const float* __restrict__ attn_in_b,
    const float* __restrict__ attn_out_w, const float* __restrict__ attn_out_b,
    const float* __restrict__ u_ws, float* __restrict__ attn_out_ws) {
  __shared__ float attn[HDIM];
  const int t = threadIdx.x;
  const float* wv = attn_in_w + (size_t)(2 * HDIM + t) * HDIM;
  const float* uh = u_ws + (t >> 5) * HDIM;
  float a = attn_in_b[2 * HDIM + t];
  for (int c = 0; c < HDIM; ++c) a += wv[c] * uh[c];
  attn[t] = a;
  __syncthreads();
  float o = attn_out_b[t];
  for (int i = 0; i < HDIM; ++i) o += attn_out_w[(size_t)t * HDIM + i] * attn[i];
  attn_out_ws[t] = o;
}

// ---------------------------------------------------------------------------
// K5: logits[n] = keys[n] . attn_out  (wave-per-row, shuffle reduce)
// ---------------------------------------------------------------------------
__global__ __launch_bounds__(256) void logits_kernel(
    const _Float16* __restrict__ keys_ws, const float* __restrict__ attn_out_ws,
    float* __restrict__ out) {
  const int lane = threadIdx.x & 31;
  const int wave = threadIdx.x >> 5;
  const int n = blockIdx.x * 8 + wave;
  const _Float16* kr = keys_ws + (size_t)n * HDIM;
  float acc = 0.f;
#pragma unroll
  for (int i = 0; i < 8; ++i) {
    const int c = lane + i * 32;
    acc += (float)kr[c] * attn_out_ws[c];
  }
#pragma unroll
  for (int off = 16; off > 0; off >>= 1) acc += __shfl_down(acc, off, 32);
  if (lane == 0) out[NTOT + n] = acc;  // logits = second output
}

// ---------------------------------------------------------------------------
// K6: global softmax stats over logits (single block)
// ---------------------------------------------------------------------------
__global__ __launch_bounds__(1024) void logit_stats_kernel(
    const float* __restrict__ out, float* __restrict__ stats_ws) {
  __shared__ float red[1024];
  const int t = threadIdx.x;
  const float* lg = out + NTOT;
  float mx = -3.0e38f;
  for (int n = t; n < NTOT; n += 1024) mx = fmaxf(mx, lg[n]);
  red[t] = mx; __syncthreads();
  for (int off = 512; off > 0; off >>= 1) {
    if (t < off) red[t] = fmaxf(red[t], red[t + off]);
    __syncthreads();
  }
  mx = red[0]; __syncthreads();
  float sum = 0.f;
  for (int n = t; n < NTOT; n += 1024) sum += __expf(lg[n] - mx);
  red[t] = sum; __syncthreads();
  for (int off = 512; off > 0; off >>= 1) {
    if (t < off) red[t] += red[t + off];
    __syncthreads();
  }
  if (t == 0) { stats_ws[16] = mx; stats_ws[17] = red[0]; }
}

// K7: probs = softmax(logits)
__global__ __launch_bounds__(256) void probs_kernel(
    const float* __restrict__ stats_ws, float* __restrict__ out) {
  const int n = blockIdx.x * 256 + threadIdx.x;
  const float inv = 1.0f / stats_ws[17];
  out[n] = __expf(out[n + NTOT] - stats_ws[16]) * inv;
}

// ---------------------------------------------------------------------------
extern "C" void kernel_launch(void* const* d_in, const int* in_sizes, int n_in,
                              void* d_out, int out_size, void* d_ws, size_t ws_size,
                              hipStream_t stream) {
  const float* qubit_emb  = (const float*)d_in[0];
  const float* qpu_emb    = (const float*)d_in[1];
  const float* gate_emb   = (const float*)d_in[2];
  const float* all_emb    = (const float*)d_in[3];
  const float* time_table = (const float*)d_in[4];
  const float* map_w1 = (const float*)d_in[5];
  const float* map_b1 = (const float*)d_in[6];
  const float* map_w2 = (const float*)d_in[7];
  const float* map_b2 = (const float*)d_in[8];
  const float* sch_w1 = (const float*)d_in[9];
  const float* sch_b1 = (const float*)d_in[10];
  const float* sch_w2 = (const float*)d_in[11];
  const float* sch_b2 = (const float*)d_in[12];
  const float* qg_w1 = (const float*)d_in[13];
  const float* qg_b1 = (const float*)d_in[14];
  const float* qg_w2 = (const float*)d_in[15];
  const float* qg_b2 = (const float*)d_in[16];
  const float* attn_in_w  = (const float*)d_in[17];
  const float* attn_in_b  = (const float*)d_in[18];
  const float* attn_out_w = (const float*)d_in[19];
  const float* attn_out_b = (const float*)d_in[20];
  const int* map_qubit  = (const int*)d_in[21];
  const int* map_qpu    = (const int*)d_in[22];
  const int* sched_gate = (const int*)d_in[23];
  const int* sched_time = (const int*)d_in[24];
  float* out = (float*)d_out;  // [0,NTOT)=probs, [NTOT,2*NTOT)=logits

  // workspace layout (bytes)
  char* ws = (char*)d_ws;
  float*    q_ws        = (float*)(ws + 0);               // 256 f32
  float*    stats_ws    = (float*)(ws + 1024);            // 32 f32
  float*    attn_out_ws = (float*)(ws + 2048);            // 256 f32
  float*    u_ws        = (float*)(ws + 4096);            // 8*256 f32
  float*    scores_ws   = (float*)(ws + 16384);           // 4 MB
  float*    wnorm_ws    = (float*)(ws + 16384 + 4194304); // 4 MB
  _Float16* wh_ws       = (_Float16*)(ws + 16384 + 2 * 4194304);      // 1 MB slot
  _Float16* keys_ws     = (_Float16*)(ws + 16384 + 2 * 4194304 + 1048576); // 64 MB

  convert_weights_kernel<<<(WH_TOTAL + 255) / 256, 256, 0, stream>>>(
      map_w1, sch_w1, map_w2, sch_w2, attn_in_w, wh_ws);

  query_kernel<<<1, 256, 0, stream>>>(all_emb, qg_w1, qg_b1, qg_w2, qg_b2,
                                      attn_in_w, attn_in_b, q_ws);

  keys_scores_kernel<<<NTOT / 16, 256, 0, stream>>>(
      qubit_emb, qpu_emb, gate_emb, time_table,
      map_b1, map_b2, sch_b1, sch_b2, attn_in_b, wh_ws,
      map_qubit, map_qpu, sched_gate, sched_time,
      q_ws, keys_ws, scores_ws);

  head_softmax_kernel<<<8, 256, 0, stream>>>(scores_ws, stats_ws, wnorm_ws, u_ws);

  weighted_sum_kernel<<<NTOT / 256, 256, 0, stream>>>(wnorm_ws, keys_ws, u_ws);

  attn_out_kernel<<<1, 256, 0, stream>>>(attn_in_w, attn_in_b, attn_out_w,
                                         attn_out_b, u_ws, attn_out_ws);

  logits_kernel<<<NTOT / 8, 256, 0, stream>>>(keys_ws, attn_out_ws, out);

  logit_stats_kernel<<<1, 1024, 0, stream>>>(out, stats_ws);

  probs_kernel<<<NTOT / 256, 256, 0, stream>>>(stats_ws, out);
}
